// GATv2Block_60550448939184
// MI455X (gfx1250) — matrix operand order repaired and checked by reading, not compiled, not run on hardware
//
#include <hip/hip_runtime.h>
#include <hip/hip_bf16.h>

// ---------------- problem constants (from reference) ----------------
#define N_NODES 50000
#define F_IN    128
#define H_HEADS 4
#define C_CH    32
#define HC      128          // H*C
#define E_EDGES 800000
#define ET      (E_EDGES + N_NODES)   // with self loops = 850000
#define NEG_SLOPE 0.2f

typedef __attribute__((ext_vector_type(16))) _Float16 v16h;
typedef __attribute__((ext_vector_type(8)))  _Float16 v8h;
typedef __attribute__((ext_vector_type(8)))  float    v8f;

union V16 { v16h v; v8h h[2]; };
union V8F { v8f v; float f[8]; };

// ---- order-preserving uint key for float atomic max ----
__device__ __forceinline__ unsigned enc_key(float f) {
    unsigned u = __float_as_uint(f);
    return (u & 0x80000000u) ? ~u : (u | 0x80000000u);
}
__device__ __forceinline__ float dec_key(unsigned e) {
    return (e & 0x80000000u) ? __uint_as_float(e & 0x7fffffffu)
                             : __uint_as_float(~e);
}

// ---------------- kernels ----------------

// zero max-keys, denominators, and the output accumulator region of d_out
__global__ void k_init(unsigned* __restrict__ mx, float* __restrict__ den,
                       float* __restrict__ outacc) {
    int i = blockIdx.x * blockDim.x + threadIdx.x;
    if (i < N_NODES * H_HEADS) { mx[i] = 0u; den[i] = 0.0f; }
    if (i < N_NODES * HC)      { outacc[i] = 0.0f; }
}

// x -> f16; Wl,Wr -> f16 transposed (col-major: WT[n*128+k] = W[k*128+n])
__global__ void k_convert(const float* __restrict__ x,
                          const float* __restrict__ Wl,
                          const float* __restrict__ Wr,
                          _Float16* __restrict__ xh,
                          _Float16* __restrict__ wlt,
                          _Float16* __restrict__ wrt) {
    int i = blockIdx.x * blockDim.x + threadIdx.x;
    if (i < N_NODES * F_IN) xh[i] = (_Float16)x[i];
    if (i < F_IN * HC) {
        int n = i >> 7;      // output column
        int k = i & 127;     // input feature
        wlt[i] = (_Float16)Wl[k * HC + n];
        wrt[i] = (_Float16)Wr[k * HC + n];
    }
}

// Fused dual projection: one wave computes one 16x16 tile of BOTH xl and xr.
// A (x rows, f16) is loaded once per K-chunk and feeds two WMMAs (Wl and Wr),
// halving streamed-A traffic vs. two separate GEMM launches. 8 WMMAs/wave.
//
// A layout (16-bit A 16x32): lanes 0-15 hold M=0..15, K = kb+{0..15} (contiguous, same order as B);
// lanes 16-31 hold K = kb+{16..31}.
// B layout (32x16): lane holds column N=lane&15; lanes 0-15 K=kb+0..15,
// lanes 16-31 K=kb+16..31 (contiguous in pre-transposed WT).
__global__ __launch_bounds__(256)
void k_gemm_wmma_dual(const _Float16* __restrict__ xh,
                      const _Float16* __restrict__ wlt,
                      const _Float16* __restrict__ wrt,
                      float* __restrict__ xl,
                      float* __restrict__ xr) {
    const int lane    = threadIdx.x & 31;
    const int wave    = threadIdx.x >> 5;
    const int hi      = lane >> 4;       // half-wave select
    const int nn      = lane & 15;       // row (A) / col (B) index
    const int rowbase = blockIdx.x * 16;
    const int colbase = wave * 16;       // 8 waves cover 128 output cols

    const _Float16* arow  = xh  + (size_t)(rowbase + nn) * F_IN;
    const _Float16* bcoll = wlt + (size_t)(colbase + nn) * F_IN;
    const _Float16* bcolr = wrt + (size_t)(colbase + nn) * F_IN;

    v8f accl = {};
    v8f accr = {};
#pragma unroll
    for (int kk = 0; kk < 4; ++kk) {
        const int kb = kk * 32;
        V16 a;
        a.v = *(const v16h*)(arow + kb + hi * 16);         // K kb+{0..15}/{16..31}
        v16h bl = *(const v16h*)(bcoll + kb + hi * 16);    // K kb+{0..15}/{16..31}
        v16h br = *(const v16h*)(bcolr + kb + hi * 16);
        accl = __builtin_amdgcn_wmma_f32_16x16x32_f16(
                   false, a.v, false, bl, (short)0, accl, false, false);
        accr = __builtin_amdgcn_wmma_f32_16x16x32_f16(
                   false, a.v, false, br, (short)0, accr, false, false);
    }
    V8F rl; rl.v = accl;
    V8F rr; rr.v = accr;
#pragma unroll
    for (int m = 0; m < 8; ++m) {
        const size_t o = (size_t)(rowbase + m + hi * 8) * HC + colbase + nn;
        xl[o] = rl.f[m];
        xr[o] = rr.f[m];
    }
}

// per (edge, head): logits = att . leakyrelu(xl[src]+xr[dst]); atomicMax key
__global__ void k_logits(const long long* __restrict__ ei,
                         const float* __restrict__ xl,
                         const float* __restrict__ xr,
                         const float* __restrict__ att,
                         float* __restrict__ lg,
                         unsigned* __restrict__ mx) {
    int t = blockIdx.x * blockDim.x + threadIdx.x;
    if (t >= ET * H_HEADS) return;
    int e = t >> 2, h = t & 3;
    int s, d;
    if (e < E_EDGES) {
        __builtin_prefetch(ei + e + 2048, 0, 0);           // global_prefetch_b8
        s = (int)ei[e];
        d = (int)ei[E_EDGES + e];
    } else {
        s = d = e - E_EDGES;                               // self loop
    }
    const float4* pl = (const float4*)(xl + (size_t)s * HC + h * C_CH);
    const float4* pr = (const float4*)(xr + (size_t)d * HC + h * C_CH);
    const float4* pa = (const float4*)(att + h * C_CH);
    float acc = 0.0f;
#pragma unroll
    for (int j = 0; j < 8; ++j) {
        float4 a = pl[j], b = pr[j], w = pa[j];
        float v;
        v = a.x + b.x; acc += w.x * (v > 0.0f ? v : NEG_SLOPE * v);
        v = a.y + b.y; acc += w.y * (v > 0.0f ? v : NEG_SLOPE * v);
        v = a.z + b.z; acc += w.z * (v > 0.0f ? v : NEG_SLOPE * v);
        v = a.w + b.w; acc += w.w * (v > 0.0f ? v : NEG_SLOPE * v);
    }
    lg[t] = acc;
    atomicMax(mx + d * H_HEADS + h, enc_key(acc));
}

// per (edge, head): ex = exp(lg - max[dst]); den[dst] += ex (overwrite lg with ex)
__global__ void k_expsum(const long long* __restrict__ ei,
                         float* __restrict__ lg,
                         const unsigned* __restrict__ mx,
                         float* __restrict__ den) {
    int t = blockIdx.x * blockDim.x + threadIdx.x;
    if (t >= ET * H_HEADS) return;
    int e = t >> 2, h = t & 3;
    int d = (e < E_EDGES) ? (int)ei[E_EDGES + e] : (e - E_EDGES);
    float m  = dec_key(mx[d * H_HEADS + h]);
    float ex = __expf(lg[t] - m);
    lg[t] = ex;
    atomicAdd(den + d * H_HEADS + h, ex);
}

// per (edge, head): alpha = ex/den[dst]; emit alpha; scatter alpha*xl[src] into out
__global__ void k_alpha_agg(const long long* __restrict__ ei,
                            const float* __restrict__ ex,
                            const float* __restrict__ den,
                            const float* __restrict__ xl,
                            float* __restrict__ alpha_out,
                            float* __restrict__ outacc) {
    int t = blockIdx.x * blockDim.x + threadIdx.x;
    if (t >= ET * H_HEADS) return;
    int e = t >> 2, h = t & 3;
    int s, d;
    if (e < E_EDGES) { s = (int)ei[e]; d = (int)ei[E_EDGES + e]; }
    else             { s = d = e - E_EDGES; }
    float alpha = ex[t] / den[d * H_HEADS + h];
    alpha_out[t] = alpha;
    const float* src = xl + (size_t)s * HC + h * C_CH;
    float* dst       = outacc + (size_t)d * HC + h * C_CH;
#pragma unroll
    for (int c = 0; c < C_CH; ++c)
        atomicAdd(dst + c, alpha * src[c]);   // global_atomic_add_f32 (L2-resident)
}

// out = relu(out + bias), in place on d_out
__global__ void k_bias_relu(float* __restrict__ outacc,
                            const float* __restrict__ bias) {
    int i = blockIdx.x * blockDim.x + threadIdx.x;
    if (i >= N_NODES * HC) return;
    float v = outacc[i] + bias[i & 127];
    outacc[i] = v > 0.0f ? v : 0.0f;
}

// ---------------- launcher ----------------
extern "C" void kernel_launch(void* const* d_in, const int* in_sizes, int n_in,
                              void* d_out, int out_size, void* d_ws, size_t ws_size,
                              hipStream_t stream) {
    const float*     x    = (const float*)d_in[0];
    const long long* ei   = (const long long*)d_in[1];  // int64 edge_index [2,E]
    const float*     Wl   = (const float*)d_in[2];
    const float*     Wr   = (const float*)d_in[3];
    const float*     att  = (const float*)d_in[4];
    const float*     bias = (const float*)d_in[5];

    float* out_nodes = (float*)d_out;                         // N*HC floats
    float* out_alpha = (float*)d_out + (size_t)N_NODES * HC;  // ET*H floats

    // workspace carve-up (all offsets 256B-aligned)
    char* ws = (char*)d_ws;
    _Float16* xh  = (_Float16*)(ws);                              // 12,800,000 B
    _Float16* wlt = (_Float16*)(ws + 12800000);                   //     32,768 B
    _Float16* wrt = (_Float16*)(ws + 12832768);                   //     32,768 B
    float*    xl  = (float*)   (ws + 12865536);                   // 25,600,000 B
    float*    xr  = (float*)   (ws + 38465536);                   // 25,600,000 B
    float*    lg  = (float*)   (ws + 64065536);                   // 13,600,000 B
    unsigned* mx  = (unsigned*)(ws + 77665536);                   //    800,000 B
    float*    den = (float*)   (ws + 78465536);                   //    800,000 B
    // total: ~79.3 MB

    const int B = 256;
    const int g_nodes = (N_NODES * HC + B - 1) / B;      // 25000
    const int g_edges = (ET * H_HEADS + B - 1) / B;      // 13282

    k_init   <<<g_nodes, B, 0, stream>>>(mx, den, out_nodes);
    k_convert<<<g_nodes, B, 0, stream>>>(x, Wl, Wr, xh, wlt, wrt);

    // xl = x @ Wl and xr = x @ Wr fused: A loaded once, 8 WMMAs per wave
    k_gemm_wmma_dual<<<N_NODES / 16, B, 0, stream>>>(xh, wlt, wrt, xl, xr);

    k_logits   <<<g_edges, B, 0, stream>>>(ei, xl, xr, att, lg, mx);
    k_expsum   <<<g_edges, B, 0, stream>>>(ei, lg, mx, den);
    k_alpha_agg<<<g_edges, B, 0, stream>>>(ei, lg, den, xl, out_alpha, out_nodes);
    k_bias_relu<<<g_nodes, B, 0, stream>>>(out_nodes, bias);
}
